// ClusterForecasting_25417616458263
// MI455X (gfx1250) — matrix operand order repaired and checked by reading, not compiled, not run on hardware
//
#include <hip/hip_runtime.h>
#include <math.h>

// ---------------------------------------------------------------------------
// MI455X (gfx1250, wave32) implementation.
//  k1: embed  h = x@w_emb + b_emb                     -> H   [64][128][64]
//  k2: attention (WMMA f16 projections + online softmax + WMMA wo, residual)
//  k3: FFN (WMMA, chunked over DF=256, residual)
//  k4: down-projection -> x_rec (d_out tail + scratch)
//  k5: fused pairwise-cost + soft-DTW: one wave32 per pair, DP state in VGPRs,
//      neighbor exchange via shfl_up, xb staged via global_load_async_to_lds
//  k6: top-5 smallest dist per row (== top_k of softmax(-dist)) + mean loss
// ---------------------------------------------------------------------------

typedef _Float16 half16 __attribute__((ext_vector_type(16)));
typedef float    float8 __attribute__((ext_vector_type(8)));

#define BIGV 100000000.0f
#define GAMMA_F 0.1f

// --- WMMA fragment loaders (wave32 layouts from CDNA5 ISA 7.12.2) -----------
__device__ __forceinline__ half16 load_a_frag(const float* A, int lda, int mr,
                                              int kb, int kRem, int lane) {
  int lo = lane & 15, hi = lane >> 4;
  half16 a;
#pragma unroll
  for (int p = 0; p < 16; ++p) {
    int kk = ((p < 8) ? 0 : 16) + hi * 8 + (p & 7);
    float v = (kk < kRem) ? A[(mr + lo) * lda + kb + kk] : 0.0f;
    a[p] = (_Float16)v;
  }
  return a;
}
__device__ __forceinline__ half16 load_b_frag(const float* W, int ldw, int kb,
                                              int nc, int kRem, int lane) {
  int lo = lane & 15, hi = lane >> 4;
  half16 b;
#pragma unroll
  for (int p = 0; p < 16; ++p) {
    int kk = ((p < 8) ? 0 : 16) + hi * 8 + (p & 7);
    float v = (kk < kRem) ? W[(kb + kk) * ldw + nc + lo] : 0.0f;
    b[p] = (_Float16)v;
  }
  return b;
}

// C[128xN] = A[128xK] @ W[KxN] (+bias,relu) or C += ...; 128 threads = 4 waves
__device__ void gemm128(const float* A, int lda, const float* W, int ldw,
                        float* C, int ldc, int K, int N, const float* bias,
                        int biasOff, bool relu, bool accumulate, int tid) {
  int wave = tid >> 5, lane = tid & 31;
  int ntiles = N >> 4;
  int total = 8 * ntiles;
  for (int tile = wave; tile < total; tile += 4) {
    int mt = tile / ntiles, nt = tile - mt * ntiles;
    int mr = mt * 16, nc = nt * 16;
    float8 acc = {};
    for (int kb = 0; kb < K; kb += 32) {
      half16 af = load_a_frag(A, lda, mr, kb, K - kb, lane);
      half16 bf = load_b_frag(W, ldw, kb, nc, K - kb, lane);
      acc = __builtin_amdgcn_wmma_f32_16x16x32_f16(false, af, false, bf,
                                                   (short)0, acc, false, false);
    }
    int lo = lane & 15, hi = lane >> 4;
#pragma unroll
    for (int r = 0; r < 8; ++r) {
      int row = mr + r + hi * 8;
      int col = nc + lo;
      float v = acc[r];
      if (bias) v += bias[biasOff + col];
      if (relu) v = fmaxf(v, 0.0f);
      if (accumulate) C[row * ldc + col] += v;
      else            C[row * ldc + col] = v;
    }
  }
}

// --- k1: embedding (K=2, pure VALU) ----------------------------------------
__global__ __launch_bounds__(128) void k_embed(const float* __restrict__ x,
                                               const float* __restrict__ w_emb,
                                               const float* __restrict__ b_emb,
                                               float* __restrict__ H) {
  int b = blockIdx.x, t = threadIdx.x;
  float x0 = x[(b * 128 + t) * 2 + 0];
  float x1 = x[(b * 128 + t) * 2 + 1];
  float* hr = H + (b * 128 + t) * 64;
#pragma unroll 4
  for (int d = 0; d < 64; ++d)
    hr[d] = fmaf(x0, w_emb[d], fmaf(x1, w_emb[64 + d], b_emb[d]));
}

// --- k2: attention ----------------------------------------------------------
__global__ __launch_bounds__(128) void k_attn(float* __restrict__ H,
                                              const float* __restrict__ wq,
                                              const float* __restrict__ wk,
                                              const float* __restrict__ wv,
                                              const float* __restrict__ wo) {
  __shared__ float Qh[128 * 16];
  __shared__ float Kh[128 * 16];
  __shared__ float Vh[128 * 16];
  __shared__ float Acc[128 * 64];
  int b = blockIdx.x, tid = threadIdx.x;
  float* Hb = H + b * 128 * 64;
  for (int i = tid; i < 128 * 64; i += 128) Acc[i] = Hb[i];   // residual base

  for (int h = 0; h < 4; ++h) {
    __syncthreads();
    gemm128(Hb, 64, wq + h * 16, 64, Qh, 16, 64, 16, nullptr, 0, false, false, tid);
    gemm128(Hb, 64, wk + h * 16, 64, Kh, 16, 64, 16, nullptr, 0, false, false, tid);
    gemm128(Hb, 64, wv + h * 16, 64, Vh, 16, 64, 16, nullptr, 0, false, false, tid);
    __syncthreads();

    // online-softmax attention, one query row per thread (s = tid)
    float q[16];
#pragma unroll
    for (int d = 0; d < 16; ++d) q[d] = Qh[tid * 16 + d];
    float m = -3.4e38f, l = 0.0f, o[16];
#pragma unroll
    for (int d = 0; d < 16; ++d) o[d] = 0.0f;
    for (int t = 0; t < 128; ++t) {
      float sc = 0.0f;
#pragma unroll
      for (int d = 0; d < 16; ++d) sc = fmaf(q[d], Kh[t * 16 + d], sc);
      sc *= 0.25f;  // 1/sqrt(16)
      float mn = fmaxf(m, sc);
      float corr = __expf(m - mn);
      float p = __expf(sc - mn);
      l = l * corr + p;
#pragma unroll
      for (int d = 0; d < 16; ++d) o[d] = fmaf(o[d], corr, p * Vh[t * 16 + d]);
      m = mn;
    }
    float invl = 1.0f / l;
#pragma unroll
    for (int d = 0; d < 16; ++d) Qh[tid * 16 + d] = o[d] * invl;  // reuse Qh
    __syncthreads();
    // Acc += out_h @ wo[h*16 : h*16+16, :]
    gemm128(Qh, 16, wo + (h * 16) * 64, 64, Acc, 64, 16, 64, nullptr, 0, false,
            true, tid);
  }
  __syncthreads();
  for (int i = tid; i < 128 * 64; i += 128) Hb[i] = Acc[i];
}

// --- k3: FFN ----------------------------------------------------------------
__global__ __launch_bounds__(128) void k_ffn(float* __restrict__ H,
                                             const float* __restrict__ w1,
                                             const float* __restrict__ b1,
                                             const float* __restrict__ w2,
                                             const float* __restrict__ b2) {
  __shared__ float Tbuf[128 * 64];
  __shared__ float Acc[128 * 64];
  int b = blockIdx.x, tid = threadIdx.x;
  float* Hb = H + b * 128 * 64;
  for (int i = tid; i < 128 * 64; i += 128) Acc[i] = Hb[i] + b2[i & 63];

  for (int c = 0; c < 4; ++c) {  // DF=256 in 4 chunks of 64
    __syncthreads();
    gemm128(Hb, 64, w1 + c * 64, 256, Tbuf, 64, 64, 64, b1, c * 64, true, false,
            tid);
    __syncthreads();
    gemm128(Tbuf, 64, w2 + (c * 64) * 64, 64, Acc, 64, 64, 64, nullptr, 0,
            false, true, tid);
  }
  __syncthreads();
  for (int i = tid; i < 128 * 64; i += 128) Hb[i] = Acc[i];
}

// --- k4: down-projection ----------------------------------------------------
__global__ __launch_bounds__(128) void k_down(const float* __restrict__ H,
                                              const float* __restrict__ w_down,
                                              const float* __restrict__ b_down,
                                              float* __restrict__ XR,
                                              float* __restrict__ out_xrec) {
  int b = blockIdx.x, t = threadIdx.x;
  const float* hr = H + (b * 128 + t) * 64;
  float a0 = b_down[0], a1 = b_down[1];
#pragma unroll 4
  for (int d = 0; d < 64; ++d) {
    float hv = hr[d];
    a0 = fmaf(hv, w_down[d * 2 + 0], a0);
    a1 = fmaf(hv, w_down[d * 2 + 1], a1);
  }
  int idx = (b * 128 + t) * 2;
  XR[idx] = a0;       XR[idx + 1] = a1;
  out_xrec[idx] = a0; out_xrec[idx + 1] = a1;
}

// --- k5: fused pairwise cost + soft-DTW, one wave32 per pair ----------------
// Lane L owns DP rows i = 4L+1 .. 4L+4; diagonals k-1 (rp*) and k-2 (rpp*)
// live in registers; boundary cells come from lane L-1 via shfl_up.
__device__ __forceinline__ float softmin_cell(float aa, float bb, float cc,
                                              float d) {
  float mn = fminf(aa, fminf(bb, cc));
  float s = __expf((mn - aa) * (1.0f / GAMMA_F)) +
            __expf((mn - bb) * (1.0f / GAMMA_F)) +
            __expf((mn - cc) * (1.0f / GAMMA_F));
  return d + mn - GAMMA_F * __logf(s);
}

__global__ __launch_bounds__(32) void k_dtw(const float* __restrict__ XR,
                                            float* __restrict__ DIST) {
  __shared__ __align__(16) float xb[256];
  int blk = blockIdx.x;
  int a = blk >> 6, b = blk & 63;
  int lane = threadIdx.x;  // wave32: one wave per block

  // Stage xb (1KB) into LDS via CDNA5 async-to-LDS path (ASYNCcnt).
  // 2 x (32 lanes x 16B); offset:512 applies to both LDS and global address.
  {
    unsigned lds_addr = (unsigned)(size_t)xb + (unsigned)lane * 16u;
    unsigned voff = (unsigned)lane * 16u;
    unsigned long long gbase = (unsigned long long)(size_t)(XR + b * 256);
    asm volatile(
        "global_load_async_to_lds_b128 %0, %1, %2\n\t"
        "global_load_async_to_lds_b128 %0, %1, %2 offset:512\n\t"
        "s_wait_asynccnt 0"
        :
        : "v"(lds_addr), "v"(voff), "s"(gbase)
        : "memory");
  }

  // xa rows for this lane (s = 4*lane .. 4*lane+3) -> registers
  float xr[8];
  {
    const float4* ga = (const float4*)(XR + a * 256);
    float4 t0 = ga[lane * 2 + 0];
    float4 t1 = ga[lane * 2 + 1];
    xr[0] = t0.x; xr[1] = t0.y; xr[2] = t0.z; xr[3] = t0.w;
    xr[4] = t1.x; xr[5] = t1.y; xr[6] = t1.z; xr[7] = t1.w;
  }
  const float2* xb2 = (const float2*)xb;

  float rp0 = BIGV, rp1 = BIGV, rp2 = BIGV, rp3 = BIGV;      // diag k-1
  float rpp0 = BIGV, rpp1 = BIGV, rpp2 = BIGV, rpp3 = BIGV;  // diag k-2
  int ibase = 4 * lane + 1;

  for (int k = 2; k <= 256; ++k) {
    float up_rp = __shfl_up(rp3, 1, 32);    // R[ibase-1, k-ibase]
    float up_rpp = __shfl_up(rpp3, 1, 32);  // R[ibase-1, k-ibase-1]
    if (lane == 0) {
      up_rp = BIGV;                         // R[0, k-1] (k>=2)
      up_rpp = (k == 2) ? 0.0f : BIGV;      // R[0, k-2]
    }
    float n0 = BIGV, n1 = BIGV, n2 = BIGV, n3 = BIGV;
    int j0 = k - ibase;  // j for q=0; decreases by 1 per q
#pragma unroll
    for (int q = 0; q < 4; ++q) {
      int j = j0 - q;
      if (j >= 1 && j <= 128) {
        float aa = (q == 0) ? up_rp : ((q == 1) ? rp0 : ((q == 2) ? rp1 : rp2));
        float bb = (q == 0) ? rp0 : ((q == 1) ? rp1 : ((q == 2) ? rp2 : rp3));
        float cc = (q == 0) ? up_rpp : ((q == 1) ? rpp0 : ((q == 2) ? rpp1 : rpp2));
        float2 xv = xb2[j - 1];
        float dx0 = xr[2 * q] - xv.x;
        float dx1 = xr[2 * q + 1] - xv.y;
        float val = softmin_cell(aa, bb, cc, dx0 * dx0 + dx1 * dx1);
        if (q == 0) n0 = val; else if (q == 1) n1 = val;
        else if (q == 2) n2 = val; else n3 = val;
      }
    }
    rpp0 = rp0; rpp1 = rp1; rpp2 = rp2; rpp3 = rp3;
    rp0 = n0; rp1 = n1; rp2 = n2; rp3 = n3;
  }
  if (lane == 31) DIST[blk] = rp3;  // R[T,T] (i=128, j=128 at k=256)
}

// --- k6: top-5 (== top_k of softmax(-dist)) + loss --------------------------
__global__ __launch_bounds__(64) void k_topk(const float* __restrict__ DIST,
                                             float* __restrict__ out) {
  __shared__ float red[64];
  int a = threadIdx.x;
  float row[64];
  for (int j = 0; j < 64; ++j) row[j] = DIST[a * 64 + j];
  unsigned long long used = 0ull;
  float sum5 = 0.0f;
  for (int q = 0; q < 5; ++q) {
    float bestv = 3.4e38f; int besti = 0;
    for (int j = 0; j < 64; ++j) {
      if (!((used >> j) & 1ull) && row[j] < bestv) { bestv = row[j]; besti = j; }
    }
    used |= (1ull << besti);
    sum5 += bestv;
    out[1 + a * 5 + q] = (float)besti;  // k_nearest stored as float
  }
  red[a] = sum5;
  __syncthreads();
  if (a == 0) {
    float tot = 0.0f;
    for (int i = 0; i < 64; ++i) tot += red[i];
    out[0] = tot / 320.0f;  // mean over (64,5)
  }
}

// ---------------------------------------------------------------------------
extern "C" void kernel_launch(void* const* d_in, const int* in_sizes, int n_in,
                              void* d_out, int out_size, void* d_ws,
                              size_t ws_size, hipStream_t stream) {
  (void)in_sizes; (void)n_in; (void)out_size; (void)ws_size;
  const float* x      = (const float*)d_in[0];
  const float* w_emb  = (const float*)d_in[1];
  const float* b_emb  = (const float*)d_in[2];
  const float* wq     = (const float*)d_in[3];
  const float* wk     = (const float*)d_in[4];
  const float* wv     = (const float*)d_in[5];
  const float* wo     = (const float*)d_in[6];
  const float* w1     = (const float*)d_in[7];
  const float* b1     = (const float*)d_in[8];
  const float* w2     = (const float*)d_in[9];
  const float* b2     = (const float*)d_in[10];
  const float* w_down = (const float*)d_in[11];
  const float* b_down = (const float*)d_in[12];

  float* H    = (float*)d_ws;                 // [64][128][64]
  float* XR   = H + 64 * 128 * 64;            // [64][128][2]
  float* DIST = XR + 64 * 128 * 2;            // [64][64]
  float* outF = (float*)d_out;                // [1 + 320 + 16384]

  k_embed<<<64, 128, 0, stream>>>(x, w_emb, b_emb, H);
  k_attn <<<64, 128, 0, stream>>>(H, wq, wk, wv, wo);
  k_ffn  <<<64, 128, 0, stream>>>(H, w1, b1, w2, b2);
  k_down <<<64, 128, 0, stream>>>(H, w_down, b_down, XR, outF + 1 + 320);
  k_dtw  <<<4096, 32, 0, stream>>>(XR, DIST);
  k_topk <<<1, 64, 0, stream>>>(DIST, outF);
}
